// HRNeck_33414845563243
// MI455X (gfx1250) — matrix operand order repaired and checked by reading, not compiled, not run on hardware
//
#include <hip/hip_runtime.h>
#include <hip/hip_bf16.h>

// ---------------------------------------------------------------------------
// Problem constants (from the reference)
// ---------------------------------------------------------------------------
#define BATCH 2
#define MAPH 128
#define MAPW 128
#define N0 16384          // original points per batch
#define OUTC 256
#define OH 64
#define OW 64
#define MROWS (BATCH * OH * OW)   // 8192 GEMM rows
#define BN_EPS 1e-5f

typedef __attribute__((ext_vector_type(16))) __bf16 v16bf;
typedef __attribute__((ext_vector_type(8)))  float  v8f;

struct __attribute__((aligned(16))) U4 { unsigned int x, y, z, w; };
struct __attribute__((aligned(16))) F4 { float x, y, z, w; };

union FragBf {
    v16bf v;
    U4    q[2];
};

__device__ __forceinline__ unsigned short f2bf(float f) {
    unsigned int u = __float_as_uint(f);
    unsigned int r = u + 0x7fffu + ((u >> 16) & 1u);   // round-to-nearest-even
    return (unsigned short)(r >> 16);
}

__device__ __forceinline__ int coord_to_cell(float lx, float ly) {
    lx = fminf(fmaxf(lx, -1.0f), 1.0f);
    ly = fminf(fmaxf(ly, -1.0f), 1.0f);
    int ix = (int)rintf((lx + 1.0f) * 0.5f * (float)(MAPW - 1));
    int iy = (int)rintf((ly + 1.0f) * 0.5f * (float)(MAPH - 1));
    ix = min(max(ix, 0), MAPW - 1);
    iy = min(max(iy, 0), MAPH - 1);
    return iy * MAPW + ix;
}

// ---------------------------------------------------------------------------
// Utility: zero an f32 buffer
// ---------------------------------------------------------------------------
__global__ void zero_f32(float* __restrict__ p, int n) {
    int i = blockIdx.x * blockDim.x + threadIdx.x;
    if (i < n) p[i] = 0.0f;
}

// ---------------------------------------------------------------------------
// Scatter: per-cell point counts
// ---------------------------------------------------------------------------
__global__ void scatter_count_kernel(const float* __restrict__ loc,
                                     float* __restrict__ cnt) {
    int i = blockIdx.x * blockDim.x + threadIdx.x;   // 0 .. B*N0-1
    if (i >= BATCH * N0) return;
    int b = i >> 14;                                  // N0 = 16384
    int cell = b * (MAPH * MAPW) + coord_to_cell(loc[2 * i], loc[2 * i + 1]);
    atomicAdd(&cnt[cell], 1.0f);
}

// ---------------------------------------------------------------------------
// Scatter: token features -> per-cell feature sums (4 channels / thread)
// ---------------------------------------------------------------------------
__global__ void scatter_feat_kernel(const float* __restrict__ x,
                                    const float* __restrict__ loc,
                                    const int*   __restrict__ idx,
                                    float* __restrict__ mapsum,
                                    int nj, int cshift) {
    int cq = 1 << (cshift - 2);                       // channel quads
    long long gid = (long long)blockIdx.x * blockDim.x + threadIdx.x;
    long long total = (long long)(BATCH * N0) * cq;
    if (gid >= total) return;
    int pt = (int)(gid >> (cshift - 2));
    int c4 = ((int)gid & (cq - 1)) << 2;
    int b  = pt >> 14;
    int cell = b * (MAPH * MAPW) + coord_to_cell(loc[2 * pt], loc[2 * pt + 1]);
    int tok = idx[pt];
    const F4* f = (const F4*)(x + (((size_t)(b * nj + tok)) << cshift) + c4);
    float* dst = mapsum + (((size_t)cell) << cshift) + c4;
    F4 v = *f;
    atomicAdd(dst + 0, v.x);
    atomicAdd(dst + 1, v.y);
    atomicAdd(dst + 2, v.z);
    atomicAdd(dst + 3, v.w);
}

// ---------------------------------------------------------------------------
// Normalize cell sums by counts and convert to bf16, NHWC layout
// ---------------------------------------------------------------------------
__global__ void normalize_map_kernel(const float* __restrict__ mapsum,
                                     const float* __restrict__ cnt,
                                     unsigned short* __restrict__ mapbf,
                                     int cshift) {
    long long i = (long long)blockIdx.x * blockDim.x + threadIdx.x;
    long long total = ((long long)BATCH * MAPH * MAPW) << cshift;
    if (i >= total) return;
    int cell = (int)(i >> cshift);
    float v = mapsum[i] / (cnt[cell] + 1e-6f);
    mapbf[i] = f2bf(v);
}

// ---------------------------------------------------------------------------
// Pack weights OIHW f32 -> bf16 wpackT[oc][k], k = (ky*3+kx)*C + c
// (B-matrix transposed layout: each lane's fragment is two contiguous 8-k runs, 16 apart)
// ---------------------------------------------------------------------------
__global__ void pack_w_kernel(const float* __restrict__ w,
                              unsigned short* __restrict__ wpackT,
                              int cshift) {
    int C = 1 << cshift;
    int K = 9 * C;
    int oc = blockIdx.y;
    int k  = blockIdx.x * blockDim.x + threadIdx.x;
    if (k >= K) return;
    int plane = k >> cshift;              // ky*3+kx
    int c     = k & (C - 1);
    int ky = plane / 3;
    int kx = plane - ky * 3;
    float v = w[(((size_t)(oc << cshift) + c) * 9) + ky * 3 + kx];
    wpackT[(size_t)oc * K + k] = f2bf(v);
}

// ---------------------------------------------------------------------------
// Implicit-GEMM conv (3x3, stride 2, pad 1) using v_wmma_f32_16x16x32_bf16.
// One wave computes a 32x32 output tile (2x2 register tiling -> 4 independent
// accumulators fill the WMMA->WMMA hazard slots). Planes (ky,kx) are the
// fully-unrolled outer loop: validity + base pointers are hoisted, invalid
// lanes read a zero-filled scratch page, so the inner channel loop is pure
// b128 loads + WMMA.
// ---------------------------------------------------------------------------
#define WMMA_BF16(A, Bm, Cacc) \
    __builtin_amdgcn_wmma_f32_16x16x32_bf16(false, (A), false, (Bm), (short)0, (Cacc), false, false)

__global__ __launch_bounds__(128)
void conv_gemm_wmma_kernel(const unsigned short* __restrict__ mapbf,
                           const unsigned short* __restrict__ wpackT,
                           const unsigned short* __restrict__ zbuf,
                           float* __restrict__ ybuf,
                           int cshift) {
    const int C = 1 << cshift;
    const int K = 9 << cshift;
    const int lane = threadIdx.x & 31;
    const int wave = threadIdx.x >> 5;
    const int tile = blockIdx.x * 4 + wave;   // 2048 tiles: 256 (M/32) x 8 (N/32)
    const int tn = tile & 7;
    const int tm = tile >> 3;
    const int half = lane >> 4;               // K-half select (ISA A/B layout)
    const int l15  = lane & 15;

    // Two A rows (output pixels) per lane: m0 and m0+16
    const int m0  = tm * 32 + l15;
    const int m1  = m0 + 16;
    const int b0  = m0 >> 12, b1 = m1 >> 12;
    const int oy0 = (m0 >> 6) & 63, oy1 = (m1 >> 6) & 63;
    const int ox0 = m0 & 63,        ox1 = m1 & 63;

    // Two B columns (output channels) per lane: oc0 and oc0+16
    const int oc0 = tn * 32 + l15;
    const unsigned short* wr0 = wpackT + (size_t)oc0 * K;
    const unsigned short* wr1 = wr0 + (size_t)16 * K;

    v8f acc00 = {0.f,0.f,0.f,0.f,0.f,0.f,0.f,0.f};
    v8f acc01 = acc00, acc10 = acc00, acc11 = acc00;

#pragma unroll
    for (int plane = 0; plane < 9; ++plane) {
        const int ky = plane / 3;             // compile-time after unroll
        const int kx = plane - ky * 3;
        const int iy0 = oy0 * 2 - 1 + ky, ix0 = ox0 * 2 - 1 + kx;
        const int iy1 = oy1 * 2 - 1 + ky, ix1 = ox1 * 2 - 1 + kx;
        const bool va = (unsigned)iy0 < (unsigned)MAPH && (unsigned)ix0 < (unsigned)MAPW;
        const bool vb = (unsigned)iy1 < (unsigned)MAPH && (unsigned)ix1 < (unsigned)MAPW;
        // Invalid lanes read zeros from the scratch page (no exec masking).
        const unsigned short* a0p = va
            ? mapbf + ((((size_t)b0 * MAPH + iy0) * MAPW + ix0) << cshift) : zbuf;
        const unsigned short* a1p = vb
            ? mapbf + ((((size_t)b1 * MAPH + iy1) * MAPW + ix1) << cshift) : zbuf;
        const unsigned short* b0p = wr0 + ((size_t)plane << cshift);
        const unsigned short* b1p = wr1 + ((size_t)plane << cshift);

        for (int c0 = 0; c0 < C; c0 += 32) {
            FragBf a0, a1, f0, f1;
            const int ao = c0 + half * 8;     // A: halves own K {0..7,16..23}/{8..15,24..31}
            a0.q[0] = *(const U4*)(a0p + ao);
            a0.q[1] = *(const U4*)(a0p + ao + 16);
            a1.q[0] = *(const U4*)(a1p + ao);
            a1.q[1] = *(const U4*)(a1p + ao + 16);
            const int bo = c0 + half * 8;     // B: halves own K {0..7,16..23}/{8..15,24..31}
            f0.q[0] = *(const U4*)(b0p + bo);
            f0.q[1] = *(const U4*)(b0p + bo + 16);
            f1.q[0] = *(const U4*)(b1p + bo);
            f1.q[1] = *(const U4*)(b1p + bo + 16);
            acc00 = WMMA_BF16(a0.v, f0.v, acc00);
            acc01 = WMMA_BF16(a0.v, f1.v, acc01);
            acc10 = WMMA_BF16(a1.v, f0.v, acc10);
            acc11 = WMMA_BF16(a1.v, f1.v, acc11);
        }
    }

    // D 16x16 f32 layout: VGPR r -> row r + 8*half, lane&15 -> col
    const int col0 = tn * 32 + l15;
    const int col1 = col0 + 16;
    const int row0 = tm * 32 + half * 8;
    const int row1 = row0 + 16;
#pragma unroll
    for (int r = 0; r < 8; ++r) {
        ybuf[(size_t)(row0 + r) * OUTC + col0] = acc00[r];
        ybuf[(size_t)(row0 + r) * OUTC + col1] = acc01[r];
        ybuf[(size_t)(row1 + r) * OUTC + col0] = acc10[r];
        ybuf[(size_t)(row1 + r) * OUTC + col1] = acc11[r];
    }
}

// ---------------------------------------------------------------------------
// Per-channel sum / sumsq over 8192 rows (coalesced partials + atomics)
// ---------------------------------------------------------------------------
__global__ void stats_kernel(const float* __restrict__ ybuf,
                             float* __restrict__ sum,
                             float* __restrict__ sumsq) {
    int oc = threadIdx.x;                 // 256 threads = 256 channels
    int p0 = blockIdx.x * 32;             // 256 blocks x 32 rows
    float s = 0.f, s2 = 0.f;
#pragma unroll 4
    for (int r = 0; r < 32; ++r) {
        float v = ybuf[(size_t)(p0 + r) * OUTC + oc];
        s  += v;
        s2 += v * v;
    }
    atomicAdd(&sum[oc], s);
    atomicAdd(&sumsq[oc], s2);
}

// ---------------------------------------------------------------------------
// BatchNorm affine + NHWC->NCHW + accumulate into d_out
// ---------------------------------------------------------------------------
__global__ void finalize_kernel(const float* __restrict__ ybuf,
                                const float* __restrict__ sum,
                                const float* __restrict__ sumsq,
                                const float* __restrict__ gamma,
                                const float* __restrict__ beta,
                                float* __restrict__ out) {
    int i = blockIdx.x * blockDim.x + threadIdx.x;   // MROWS*256
    if (i >= MROWS * OUTC) return;
    int p  = i >> 8;
    int oc = i & 255;
    const float inv_n = 1.0f / (float)MROWS;
    float mean = sum[oc] * inv_n;
    float var  = sumsq[oc] * inv_n - mean * mean;
    float v = (ybuf[i] - mean) * rsqrtf(var + BN_EPS) * gamma[oc] + beta[oc];
    int b  = p >> 12;
    int oy = (p >> 6) & 63;
    int ox = p & 63;
    out[((((size_t)b * OUTC) + oc) * OH + oy) * OW + ox] += v;
}

// ---------------------------------------------------------------------------
// Host-side orchestration
// ---------------------------------------------------------------------------
static inline size_t align256(size_t x) { return (x + 255) & ~(size_t)255; }

extern "C" void kernel_launch(void* const* d_in, const int* in_sizes, int n_in,
                              void* d_out, int out_size, void* d_ws, size_t ws_size,
                              hipStream_t stream) {
    (void)in_sizes; (void)n_in; (void)out_size; (void)ws_size;

    const int CH[4]     = {64, 128, 256, 512};
    const int CSHIFT[4] = {6, 7, 8, 9};
    const int NJ[4]     = {16384, 4096, 1024, 256};

    // Workspace carve-out
    char* ws = (char*)d_ws;
    float* cnt = (float*)ws;                    ws += align256((size_t)BATCH * MAPH * MAPW * 4);
    float* mapsum = (float*)ws;                 ws += align256((size_t)BATCH * MAPH * MAPW * 512 * 4);
    unsigned short* mapbf = (unsigned short*)ws; ws += align256((size_t)BATCH * MAPH * MAPW * 512 * 2);
    unsigned short* wpackT = (unsigned short*)ws; ws += align256((size_t)OUTC * 9 * 512 * 2);
    float* ybuf = (float*)ws;                   ws += align256((size_t)MROWS * OUTC * 4);
    float* csum = (float*)ws;                   ws += align256((size_t)OUTC * 4);
    float* csumsq = (float*)ws;                 ws += align256((size_t)OUTC * 4);
    unsigned short* zbuf = (unsigned short*)ws; ws += align256((size_t)1024 * 2);  // zero page

    float* out = (float*)d_out;
    const int ncells = BATCH * MAPH * MAPW;     // 32768

    // Zero the output accumulator and the GEMM zero-page
    {
        int n = BATCH * OUTC * OH * OW;         // 2,097,152
        zero_f32<<<(n + 255) / 256, 256, 0, stream>>>(out, n);
        zero_f32<<<2, 256, 0, stream>>>((float*)zbuf, 512);
    }

    for (int j = 0; j < 4; ++j) {
        const float* x     = (const float*)d_in[6 * j + 0];
        const float* loc   = (const float*)d_in[6 * j + 1];
        const int*   idx   = (const int*)  d_in[6 * j + 2];
        const float* w     = (const float*)d_in[6 * j + 3];
        const float* gamma = (const float*)d_in[6 * j + 4];
        const float* beta  = (const float*)d_in[6 * j + 5];
        const int cshift = CSHIFT[j];
        const int C = CH[j];
        const int K = 9 * C;

        // Zero per-branch accumulators
        zero_f32<<<(ncells + 255) / 256, 256, 0, stream>>>(cnt, ncells);
        {
            long long n = (long long)ncells * C;
            zero_f32<<<(unsigned)((n + 255) / 256), 256, 0, stream>>>(mapsum, (int)n);
        }
        zero_f32<<<2, 256, 0, stream>>>(csum, OUTC);
        zero_f32<<<2, 256, 0, stream>>>(csumsq, OUTC);

        // token2map
        scatter_count_kernel<<<(BATCH * N0 + 255) / 256, 256, 0, stream>>>(loc, cnt);
        {
            long long n = (long long)BATCH * N0 * (C / 4);
            scatter_feat_kernel<<<(unsigned)((n + 255) / 256), 256, 0, stream>>>(
                x, loc, idx, mapsum, NJ[j], cshift);
        }
        {
            long long n = (long long)ncells * C;
            normalize_map_kernel<<<(unsigned)((n + 255) / 256), 256, 0, stream>>>(
                mapsum, cnt, mapbf, cshift);
        }

        // conv_bn
        {
            dim3 g((K + 255) / 256, OUTC);
            pack_w_kernel<<<g, 256, 0, stream>>>(w, wpackT, cshift);
        }
        // 2048 wave-tiles (256 M x 8 N, 32x32 each), 4 waves/block
        conv_gemm_wmma_kernel<<<512, 128, 0, stream>>>(mapbf, wpackT, zbuf, ybuf, cshift);
        stats_kernel<<<256, 256, 0, stream>>>(ybuf, csum, csumsq);
        finalize_kernel<<<(MROWS * OUTC + 255) / 256, 256, 0, stream>>>(
            ybuf, csum, csumsq, gamma, beta, out);
    }
}
